// GATModel_3212635537596
// MI455X (gfx1250) — matrix operand order repaired and checked by reading, not compiled, not run on hardware
//
#include <hip/hip_runtime.h>

#define N_SRC   200000
#define N_DST0  50000
#define N_DST1  10000
#define NE0     800000
#define NE1     320000
#define IN_C    128
#define HIDC    64
#define OUT_C   40
#define HEADS0  4
#define HC0     256      /* HEADS0*HIDC */
#define BN_EPS  1e-5f
#define SLOPE   0.2f

typedef __attribute__((ext_vector_type(16))) __bf16 v16bf;
typedef __attribute__((ext_vector_type(8)))  float  v8f;

__device__ __forceinline__ float lrelu(float v) { return v > 0.f ? v : v * SLOPE; }

// ---- scratch clear -------------------------------------------------------
__global__ void zero_kernel(unsigned* p, long long n) {
  long long i = blockIdx.x * (long long)blockDim.x + threadIdx.x;
  if (i < n) p[i] = 0u;
}

// ---- weight conversion: f32 row-major -> bf16 column-major ---------------
__global__ void convert_weights(const float* __restrict__ W0, const float* __restrict__ W1,
                                __bf16* __restrict__ W0t, __bf16* __restrict__ W1t) {
  int tid = blockIdx.x * blockDim.x + threadIdx.x;
  if (tid < IN_C * HC0) {                 // W0 [128][256] -> W0t [256][128]
    int k = tid / HC0, n = tid - k * HC0;
    W0t[n * IN_C + k] = (__bf16)W0[tid];
  }
  if (tid < 48 * HC0) {                   // W1 [256][40] -> W1t [48][256], zero-pad N
    int n = tid / HC0, k = tid - n * HC0;
    W1t[tid] = (__bf16)((n < OUT_C) ? W1[k * OUT_C + n] : 0.f);
  }
}

// ---- WMMA GEMM: C[M,Nout] = A[M,K](f32) * B (bf16, column-major Bt[N][K])
// one wave per 16x16 tile; K multiple of 32; M multiple of 16.
__global__ void gemm_bf16_wmma(const float* __restrict__ A, const __bf16* __restrict__ Bt,
                               float* __restrict__ C, int Mtiles, int Ntiles,
                               int K, int Nout, int ldc) {
  int wave = (int)((blockIdx.x * (long long)blockDim.x + threadIdx.x) >> 5);
  int lane = threadIdx.x & 31;
  if (wave >= Mtiles * Ntiles) return;            // wave-uniform: EXEC stays all-1s
  int mt = wave / Ntiles;
  int nt = wave - mt * Ntiles;
  int kgrp = lane >> 4;                            // K-half select (ISA 16-bit layouts)
  const float*  ap0 = A  + (size_t)(mt * 16 + (lane & 15)) * K;   // A row = lane&15
  const __bf16* bp0 = Bt + (size_t)(nt * 16 + (lane & 15)) * K;   // B col = lane&15
  v8f acc = {};
  for (int kk = 0; kk < K; kk += 32) {
    v16bf a, b;
    const float* ap = ap0 + kk + kgrp * 8;
    __builtin_prefetch(ap0 + kk + 32, 0, 3);       // global_prefetch next K-slab
#pragma unroll
    for (int j = 0; j < 8; ++j) {                  // A: elems 0..7 -> K=kgrp*8+j ; 8..15 -> +16
      a[j]     = (__bf16)ap[j];
      a[j + 8] = (__bf16)ap[j + 16];
    }
    const __bf16* bp = bp0 + kk + kgrp * 8;        // B: elems 0..7 -> K=kgrp*8+j ; 8..15 -> +16
#pragma unroll
    for (int j = 0; j < 16; ++j) b[j] = bp[j < 8 ? j : j + 8];
    acc = __builtin_amdgcn_wmma_f32_16x16x32_bf16(false, a, false, b, (short)0, acc,
                                                  false, false);
  }
  int crow = mt * 16 + (lane >> 4) * 8;            // C: VGPR r -> row (lane>>4)*8+r
  int ccol = nt * 16 + (lane & 15);
  if (ccol < Nout) {
#pragma unroll
    for (int r = 0; r < 8; ++r)
      C[(size_t)(crow + r) * ldc + ccol] = acc[r];
  }
}

// ---- attention logits, layer 0: wave per node, coalesced row read --------
__global__ void alpha0_kernel(const float* __restrict__ h0, const float* __restrict__ att_s,
                              const float* __restrict__ att_d, float* __restrict__ as0,
                              float* __restrict__ ad0) {
  int wave = (int)((blockIdx.x * (long long)blockDim.x + threadIdx.x) >> 5);
  int lane = threadIdx.x & 31;
  if (wave >= N_SRC) return;
  const float* hr = h0 + (size_t)wave * HC0;
  int c0 = lane * 8;                               // 8 lanes per head (64 ch / head)
  float ps = 0.f, pd = 0.f;
#pragma unroll
  for (int j = 0; j < 8; ++j) {
    float v = hr[c0 + j];
    ps += v * att_s[c0 + j];
    pd += v * att_d[c0 + j];
  }
#pragma unroll
  for (int m = 1; m < 8; m <<= 1) {                // reduce within 8-lane head group
    ps += __shfl_xor(ps, m, 32);
    pd += __shfl_xor(pd, m, 32);
  }
  if ((lane & 7) == 0) {
    int h = lane >> 3;
    as0[(size_t)wave * HEADS0 + h] = ps;
    if (wave < N_DST0) ad0[(size_t)wave * HEADS0 + h] = pd;
  }
}

// ---- attention logits, layer 1 -------------------------------------------
__global__ void alpha1_kernel(const float* __restrict__ h1, const float* __restrict__ att_s,
                              const float* __restrict__ att_d, float* __restrict__ as1,
                              float* __restrict__ ad1) {
  int n = blockIdx.x * blockDim.x + threadIdx.x;
  if (n >= N_DST0) return;
  const float* hr = h1 + (size_t)n * OUT_C;
  float ps = 0.f, pd = 0.f;
  for (int c = 0; c < OUT_C; ++c) {
    float v = hr[c];
    ps += v * att_s[c];
    pd += v * att_d[c];
  }
  as1[n] = ps;
  if (n < N_DST1) ad1[n] = pd;
}

// ---- CSR construction (edges + implicit self loops) ----------------------
__global__ void deg_kernel(const int* __restrict__ edst, unsigned* __restrict__ deg,
                           int E, int Etot) {
  int idx = blockIdx.x * blockDim.x + threadIdx.x;
  if (idx >= Etot) return;
  int d = (idx < E) ? edst[idx] : (idx - E);
  atomicAdd(&deg[d], 1u);
}

// single-block exclusive scan (LDS scan over 1024-chunks with running carry)
__global__ void scan_kernel(const unsigned* __restrict__ deg, unsigned* __restrict__ rowptr,
                            int n) {
  __shared__ unsigned sdata[1024];
  __shared__ unsigned carry;
  if (threadIdx.x == 0) carry = 0u;
  __syncthreads();
  for (int base = 0; base < n; base += 1024) {
    int i = base + (int)threadIdx.x;
    unsigned v = (i < n) ? deg[i] : 0u;
    sdata[threadIdx.x] = v;
    __syncthreads();
    for (int ofs = 1; ofs < 1024; ofs <<= 1) {
      unsigned t = (threadIdx.x >= (unsigned)ofs) ? sdata[threadIdx.x - ofs] : 0u;
      __syncthreads();
      sdata[threadIdx.x] += t;
      __syncthreads();
    }
    unsigned c = carry;                       // uniform read (prev iter synced)
    if (i < n) rowptr[i] = c + sdata[threadIdx.x] - v;   // exclusive prefix
    __syncthreads();
    if (threadIdx.x == 0) carry = c + sdata[1023];
    __syncthreads();
  }
  if (threadIdx.x == 0) rowptr[n] = carry;
}

__global__ void fill_kernel(const int* __restrict__ esrc, const int* __restrict__ edst,
                            const unsigned* __restrict__ rowptr, unsigned* __restrict__ cur,
                            int* __restrict__ eidx, int E, int Etot) {
  int idx = blockIdx.x * blockDim.x + threadIdx.x;
  if (idx >= Etot) return;
  int s, d;
  if (idx < E) { s = esrc[idx]; d = edst[idx]; } else { s = idx - E; d = s; }
  unsigned pos = rowptr[d] + atomicAdd(&cur[d], 1u);
  eidx[pos] = s;
}

// ---- gather-based GAT aggregation: one wave per (dst, head) --------------
// segment softmax (max + exp-sum via shuffle reduce) + weighted gather; no FP atomics.
__global__ void gat_aggregate(const unsigned* __restrict__ rowptr, const int* __restrict__ eidx,
                              const float* __restrict__ as, const float* __restrict__ ad,
                              const float* __restrict__ feat, float* __restrict__ outp,
                              int ndst, int H, int CH, int ld) {
  long long wave = (blockIdx.x * (long long)blockDim.x + threadIdx.x) >> 5;
  int lane = threadIdx.x & 31;
  if (wave >= (long long)ndst * H) return;
  int d = (int)(wave / H), h = (int)(wave - (long long)d * H);
  unsigned beg = rowptr[d], end = rowptr[d + 1];
  float adv = ad[(size_t)d * H + h];
  // pass 1: segment max
  float m = -3.4e38f;
  for (unsigned j = beg + lane; j < end; j += 32)
    m = fmaxf(m, lrelu(as[(size_t)eidx[j] * H + h] + adv));
#pragma unroll
  for (int o = 16; o > 0; o >>= 1) m = fmaxf(m, __shfl_xor(m, o, 32));
  // pass 2: exp-sum
  float z = 0.f;
  for (unsigned j = beg + lane; j < end; j += 32)
    z += __expf(lrelu(as[(size_t)eidx[j] * H + h] + adv) - m);
#pragma unroll
  for (int o = 16; o > 0; o >>= 1) z += __shfl_xor(z, o, 32);
  float invz = 1.f / z;                       // >= 1 edge (self loop) so z > 0
  // pass 3: weighted gather, lanes over channels (CH <= 64)
  float a0 = 0.f, a1 = 0.f;
  for (unsigned j = beg; j < end; ++j) {
    int s = eidx[j];
    float alpha = __expf(lrelu(as[(size_t)s * H + h] + adv) - m) * invz;
    const float* fs = feat + (size_t)s * ld + h * CH;
    if (lane < CH)      a0 += alpha * fs[lane];
    if (lane + 32 < CH) a1 += alpha * fs[lane + 32];
  }
  float* od = outp + (size_t)d * ld + h * CH;
  if (lane < CH)      od[lane] = a0;
  if (lane + 32 < CH) od[lane + 32] = a1;
}

// ---- BatchNorm (bias0 cancels analytically) ------------------------------
__global__ void bn_partial(const float* __restrict__ x, float* __restrict__ sums,
                           float* __restrict__ sumsq) {
  int ch = threadIdx.x;                        // 256 channels = 256 threads, coalesced
  int base = blockIdx.x * 64;
  float s = 0.f, s2 = 0.f;
  for (int r = 0; r < 64; ++r) {
    int row = base + r;
    if (row < N_DST0) {
      float v = x[(size_t)row * HC0 + ch];
      s += v; s2 += v * v;
    }
  }
  atomicAdd(&sums[ch], s);
  atomicAdd(&sumsq[ch], s2);
}

__global__ void bn_finalize(const float* __restrict__ sums, const float* __restrict__ sumsq,
                            const float* __restrict__ gamma, const float* __restrict__ beta,
                            float* __restrict__ scale, float* __restrict__ shift) {
  int c = threadIdx.x;
  float mu  = sums[c] * (1.f / N_DST0);
  float var = sumsq[c] * (1.f / N_DST0) - mu * mu;
  float sc  = rsqrtf(var + BN_EPS) * gamma[c];
  scale[c] = sc;
  shift[c] = beta[c] - mu * sc;
}

__global__ void bn_apply(float* __restrict__ x, const float* __restrict__ scale,
                         const float* __restrict__ shift) {
  long long i = blockIdx.x * (long long)blockDim.x + threadIdx.x;
  if (i >= (long long)N_DST0 * HC0) return;
  int ch = (int)(i & (HC0 - 1));
  float v = x[i] * scale[ch] + shift[ch];
  x[i] = v > 0.f ? v : 0.f;                    // ReLU
}

// ---- bias + log_softmax ---------------------------------------------------
__global__ void final_logsoftmax(const float* __restrict__ acc1, const float* __restrict__ bias,
                                 float* __restrict__ out) {
  int r = blockIdx.x * blockDim.x + threadIdx.x;
  if (r >= N_DST1) return;
  const float* a = acc1 + (size_t)r * OUT_C;
  float v[OUT_C];
  float vmax = -3.4e38f;
  for (int c = 0; c < OUT_C; ++c) { v[c] = a[c] + bias[c]; vmax = fmaxf(vmax, v[c]); }
  float ssum = 0.f;
  for (int c = 0; c < OUT_C; ++c) ssum += __expf(v[c] - vmax);
  float lse = vmax + __logf(ssum);
  float* o = out + (size_t)r * OUT_C;
  for (int c = 0; c < OUT_C; ++c) o[c] = v[c] - lse;
}

// ---- launcher -------------------------------------------------------------
extern "C" void kernel_launch(void* const* d_in, const int* in_sizes, int n_in,
                              void* d_out, int out_size, void* d_ws, size_t ws_size,
                              hipStream_t stream) {
  (void)in_sizes; (void)n_in; (void)out_size; (void)ws_size;
  const float* x      = (const float*)d_in[0];
  const float* W0     = (const float*)d_in[1];
  const float* atts0  = (const float*)d_in[2];
  const float* attd0  = (const float*)d_in[3];
  /* bias0 = d_in[4] cancels in BatchNorm */
  const float* gamma0 = (const float*)d_in[5];
  const float* beta0  = (const float*)d_in[6];
  const float* W1     = (const float*)d_in[7];
  const float* atts1  = (const float*)d_in[8];
  const float* attd1  = (const float*)d_in[9];
  const float* bias1  = (const float*)d_in[10];
  const int* es0 = (const int*)d_in[11];
  const int* ed0 = (const int*)d_in[12];
  const int* es1 = (const int*)d_in[13];
  const int* ed1 = (const int*)d_in[14];
  float* out = (float*)d_out;

  char* base = (char*)d_ws;
  size_t off = 0;
  auto alloc = [&](size_t bytes) -> char* {
    size_t o = (off + 255) & ~(size_t)255;
    off = o + bytes;
    return base + o;
  };
  const int ET0 = NE0 + N_DST0;   // edges + self loops, layer 0
  const int ET1 = NE1 + N_DST1;   // edges + self loops, layer 1

  __bf16* W0t = (__bf16*)alloc((size_t)HC0 * IN_C * 2);
  __bf16* W1t = (__bf16*)alloc((size_t)48 * HC0 * 2);
  float* h0   = (float*)alloc((size_t)N_SRC * HC0 * 4);
  float* as0  = (float*)alloc((size_t)N_SRC * HEADS0 * 4);
  float* ad0  = (float*)alloc((size_t)N_DST0 * HEADS0 * 4);
  float* acc0 = (float*)alloc((size_t)N_DST0 * HC0 * 4);   // layer-0 output / BN in-place
  float* h1   = (float*)alloc((size_t)N_DST0 * OUT_C * 4);
  float* as1  = (float*)alloc((size_t)N_DST0 * 4);
  float* ad1  = (float*)alloc((size_t)N_DST1 * 4);
  float* acc1 = (float*)alloc((size_t)N_DST1 * OUT_C * 4);
  float* scale = (float*)alloc(HC0 * 4);
  float* shift = (float*)alloc(HC0 * 4);
  unsigned* rowptr0 = (unsigned*)alloc((size_t)(N_DST0 + 1) * 4);
  unsigned* rowptr1 = (unsigned*)alloc((size_t)(N_DST1 + 1) * 4);
  int* eidx0 = (int*)alloc((size_t)ET0 * 4);
  int* eidx1 = (int*)alloc((size_t)ET1 * 4);
  // --- zeroed-every-launch region (contiguous -> one clear kernel) ---
  size_t zoff = (off + 255) & ~(size_t)255;
  unsigned* deg0 = (unsigned*)alloc((size_t)N_DST0 * 4);
  unsigned* cur0 = (unsigned*)alloc((size_t)N_DST0 * 4);
  unsigned* deg1 = (unsigned*)alloc((size_t)N_DST1 * 4);
  unsigned* cur1 = (unsigned*)alloc((size_t)N_DST1 * 4);
  float* sums  = (float*)alloc(HC0 * 4);
  float* sumsq = (float*)alloc(HC0 * 4);
  long long zero_words = (long long)((off - zoff + 3) / 4);
  unsigned* zptr = (unsigned*)(base + zoff);

  const int T = 256;

  zero_kernel<<<(unsigned)((zero_words + T - 1) / T), T, 0, stream>>>(zptr, zero_words);
  convert_weights<<<(IN_C * HC0 + T - 1) / T, T, 0, stream>>>(W0, W1, W0t, W1t);

  // CSR build for both layers (only depends on edge lists)
  deg_kernel<<<(ET0 + T - 1) / T, T, 0, stream>>>(ed0, deg0, NE0, ET0);
  deg_kernel<<<(ET1 + T - 1) / T, T, 0, stream>>>(ed1, deg1, NE1, ET1);
  scan_kernel<<<1, 1024, 0, stream>>>(deg0, rowptr0, N_DST0);
  scan_kernel<<<1, 1024, 0, stream>>>(deg1, rowptr1, N_DST1);
  fill_kernel<<<(ET0 + T - 1) / T, T, 0, stream>>>(es0, ed0, rowptr0, cur0, eidx0, NE0, ET0);
  fill_kernel<<<(ET1 + T - 1) / T, T, 0, stream>>>(es1, ed1, rowptr1, cur1, eidx1, NE1, ET1);

  { // GEMM0: h0 = x @ W0  (12500 x 16 tiles, one wave each)
    long long waves = (long long)(N_SRC / 16) * 16;
    gemm_bf16_wmma<<<(unsigned)((waves * 32 + T - 1) / T), T, 0, stream>>>(
        x, W0t, h0, N_SRC / 16, 16, IN_C, HC0, HC0);
  }
  { // attention logits layer 0
    long long waves = N_SRC;
    alpha0_kernel<<<(unsigned)((waves * 32 + T - 1) / T), T, 0, stream>>>(
        h0, atts0, attd0, as0, ad0);
  }
  { // layer-0 segment softmax + gather aggregation (wave per dst*head)
    long long waves = (long long)N_DST0 * HEADS0;
    gat_aggregate<<<(unsigned)((waves * 32 + T - 1) / T), T, 0, stream>>>(
        rowptr0, eidx0, as0, ad0, h0, acc0, N_DST0, HEADS0, HIDC, HC0);
  }
  // BatchNorm + ReLU (in place on acc0)
  bn_partial<<<(N_DST0 + 63) / 64, T, 0, stream>>>(acc0, sums, sumsq);
  bn_finalize<<<1, T, 0, stream>>>(sums, sumsq, gamma0, beta0, scale, shift);
  bn_apply<<<(unsigned)(((long long)N_DST0 * HC0 + T - 1) / T), T, 0, stream>>>(
      acc0, scale, shift);

  { // GEMM1: h1 = acc0 @ W1   (3125 x 3 tiles, N padded 40->48)
    long long waves = (long long)(N_DST0 / 16) * 3;
    gemm_bf16_wmma<<<(unsigned)((waves * 32 + T - 1) / T), T, 0, stream>>>(
        acc0, W1t, h1, N_DST0 / 16, 3, HC0, OUT_C, OUT_C);
  }
  alpha1_kernel<<<(N_DST0 + T - 1) / T, T, 0, stream>>>(h1, atts1, attd1, as1, ad1);
  { // layer-1 aggregation (H=1, CH=40)
    long long waves = N_DST1;
    gat_aggregate<<<(unsigned)((waves * 32 + T - 1) / T), T, 0, stream>>>(
        rowptr1, eidx1, as1, ad1, h1, acc1, N_DST1, 1, OUT_C, OUT_C);
  }
  final_logsoftmax<<<(N_DST1 + T - 1) / T, T, 0, stream>>>(acc1, bias1, out);
}
